// RecursiveAssociationNeuralNetworks_78357383348941
// MI455X (gfx1250) — matrix operand order repaired and checked by reading, not compiled, not run on hardware
//
#include <hip/hip_runtime.h>
#include <hip/hip_bf16.h>
#include <math.h>
#include <stdint.h>

// MI455X / gfx1250: wave32, WMMA 16x16x32 bf16 w/ f32 accumulate, TDM async copy.
typedef __bf16 bf16_t;
typedef __attribute__((ext_vector_type(16))) __bf16 v16bf;
typedef __attribute__((ext_vector_type(8)))  float  v8f;
typedef __attribute__((ext_vector_type(2)))  float  v2f;
typedef unsigned int v4u  __attribute__((ext_vector_type(4)));
typedef int          v8i_t __attribute__((ext_vector_type(8)));
typedef int          v4i_t __attribute__((ext_vector_type(4)));

#define L_LVL   6
#define N_NODES 16384
#define C_CH    8
#define H_DIM   128

union Pack8 { uint4 u; bf16_t h[8]; };
union Frag  { uint4 u[2]; v16bf v; };

__device__ __forceinline__ v8f wmma_bf16(v16bf a, v16bf b, v8f c) {
    return __builtin_amdgcn_wmma_f32_16x16x32_bf16(false, a, false, b, (short)0, c,
                                                   false, false);
}

__device__ __forceinline__ float fast_tanh(float x) {
#if __has_builtin(__builtin_amdgcn_tanhf)
    return __builtin_amdgcn_tanhf(x);   // v_tanh_f32 (gfx1250 TRANS op)
#else
    return tanhf(x);
#endif
}

// ---------------------------------------------------------------------------
// Pre-swizzle the three 128x128 weights (f32 row-major, W[k][n]) into the
// WMMA B-fragment lane layout (bf16). For K-step s (32 K's), tile t (16 N's):
//   lane l holds column n = 16t + (l&15); element j -> K = 32s + 8*(l>>4) + j + 8*(j>>3).
// One B fragment = 32 contiguous bytes per lane.
// ---------------------------------------------------------------------------
__global__ void ran_prep_weights(const float* __restrict__ Wg,
                                 const float* __restrict__ Wih,
                                 const float* __restrict__ Whh,
                                 bf16_t* __restrict__ wsw) {
    int tid  = blockIdx.x * blockDim.x + threadIdx.x;   // 0..3071
    int wsel = tid >> 10;
    int rem  = tid & 1023;
    int t    = rem >> 7;
    int s    = (rem >> 5) & 3;
    int l    = rem & 31;
    const float* W = (wsel == 0) ? Wg : (wsel == 1) ? Wih : Whh;
    bf16_t* out = wsw + (size_t)wsel * (128 * 128) + (((t * 4 + s) * 32 + l) * 16);
    int n     = t * 16 + (l & 15);
    int kbase = 32 * s + 8 * (l >> 4);
#pragma unroll
    for (int j = 0; j < 16; ++j) out[j] = (bf16_t)W[(kbase + j + (j >> 3) * 8) * 128 + n];
}

// ---------------------------------------------------------------------------
// One scan level, fully fused. Block = 256 threads (8 waves) = 16 nodes.
// Key identity: (a @ ch) @ W_g == a @ (ch @ W_g); we WMMA ch@W_g, then the
// 8x8 "a" contraction + masked child-max happen in-lane on the D-tile (rows
// of the D tile are the 8 children of the lane's node), packed as child-pairs
// so the contraction lowers to v_pk_fma_f32.
// ---------------------------------------------------------------------------
__launch_bounds__(256)
__global__ void ran_level_kernel(const float* __restrict__ x_l,       // [N][128]
                                 const int* __restrict__ cidx_l,      // [N][8]
                                 const unsigned char* __restrict__ cmask_l, // [N][8]
                                 const float* __restrict__ a_l,       // [N][8][8]
                                 const bf16_t* __restrict__ wsw,      // swizzled weights
                                 const float* __restrict__ b_ih,
                                 const float* __restrict__ b_hh,
                                 const float* __restrict__ h_below,   // [N][128]
                                 float* __restrict__ h_out,           // [N][128]
                                 int firstLevel) {
    __shared__ __align__(16) bf16_t ch_lds[16 * 8 * 128];   // 32 KB masked child rows
    __shared__ __align__(16) bf16_t wg_lds[128 * 128];      // 32 KB swizzled W_g
    __shared__ __align__(16) bf16_t pooled_lds[16 * 128];   //  4 KB pooled handoff
    __shared__ __align__(16) float  a_lds[16 * 64];         //  4 KB per-node a mats

    const int tid = threadIdx.x;
    const int blk = blockIdx.x;

    // ---- TDM: async-stage swizzled W_g (32 KB) global -> LDS, issued by wave 0.
#if __has_builtin(__builtin_amdgcn_tensor_load_to_lds)
    if (tid < 32) {
        unsigned lds_addr = (unsigned)(uintptr_t)(&wg_lds[0]);
        unsigned long long ga = (unsigned long long)(uintptr_t)wsw;  // W_g at offset 0
        // D# group0: count=1, lds_addr, global_addr, type=2
        v4u g0 = { 1u,
                   lds_addr,
                   (unsigned)(ga & 0xFFFFFFFFu),
                   (unsigned)(((ga >> 32) & 0x01FFFFFFu) | 0x80000000u) };
        // D# group1: data_size=8B; tensor_dim0=4096; tensor_dim1=1; tile_dim0=4096
        v8i_t g1 = { (int)(3u << 16),        // [17:16] data_size=3 (8B)
                     (int)(4096u << 16),     // [63:48] tensor_dim0[15:0]
                     (int)(1u << 16),        // [95:80] tensor_dim1[15:0]=1
                     (int)(4096u << 16),     // [127:112] tile_dim0=4096
                     0,                      // tile_dim1=0 (unused), tile_dim2=0
                     4096,                   // [191:160] tensor_dim0_stride lo
                     0, 0 };
        v4i_t gz4 = { 0, 0, 0, 0 };
        v8i_t gz8 = { 0, 0, 0, 0, 0, 0, 0, 0 };
        __builtin_amdgcn_tensor_load_to_lds(g0, g1, gz4, gz4, gz8, 0);
    }
#else
    for (int i = tid; i < 4096; i += 256)                    // fallback: 8B per lane
        ((uint2*)wg_lds)[i] = ((const uint2*)wsw)[i];
#endif

    // ---- Phase 0a: cooperative gather ch = mask * h_below[cidx] -> LDS (bf16)
    {
        int R    = tid >> 1;          // gathered row 0..127
        int half = tid & 1;           // which 64-float half of the row
        int nl   = R >> 3;            // local node 0..15
        int d    = R & 7;             // child slot
        int node = blk * 16 + nl;
        int idx  = cidx_l[node * 8 + d];
        int mk   = (int)cmask_l[node * 8 + d];
        bf16_t* dst = &ch_lds[(nl * 8 + d) * 128 + half * 64];
        if (firstLevel || !mk) {
            uint4 z = {0u, 0u, 0u, 0u};
#pragma unroll
            for (int i = 0; i < 8; ++i) ((uint4*)dst)[i] = z;
        } else {
            const float4* src = (const float4*)(h_below + (size_t)idx * 128 + half * 64);
#pragma unroll
            for (int i = 0; i < 8; ++i) {
                float4 f0 = src[2 * i], f1 = src[2 * i + 1];
                Pack8 pk;
                pk.h[0] = (bf16_t)f0.x; pk.h[1] = (bf16_t)f0.y;
                pk.h[2] = (bf16_t)f0.z; pk.h[3] = (bf16_t)f0.w;
                pk.h[4] = (bf16_t)f1.x; pk.h[5] = (bf16_t)f1.y;
                pk.h[6] = (bf16_t)f1.z; pk.h[7] = (bf16_t)f1.w;
                ((uint4*)dst)[i] = pk.u;
            }
        }
    }
    // ---- Phase 0b: cooperative copy of this block's a matrices (f32) -> LDS
    ((float4*)a_lds)[tid] = ((const float4*)(a_l + (size_t)blk * 1024))[tid];

#if __has_builtin(__builtin_amdgcn_tensor_load_to_lds)
#if __has_builtin(__builtin_amdgcn_s_wait_tensorcnt)
    if (tid < 32) __builtin_amdgcn_s_wait_tensorcnt(0);
#endif
#endif
    __syncthreads();

    const int wave = tid >> 5;      // 0..7 ; owns nodes (2w, 2w+1) in phase 1
    const int lane = tid & 31;
    const int r    = lane & 15;     // A-fragment row / output column-in-tile
    const int hi   = lane >> 4;
    const int g0n  = blk * 16 + 2 * wave;

    __builtin_prefetch(x_l + (size_t)(blk * 16 + r) * 128, 0, 3);

    // ---- Phase 1: u = ch @ W_g via WMMA. A-fragments are ch rows read
    // directly from LDS (16-bit A 16x32 layout; chunks of 8 consecutive K).
    v16bf afrag[4];
    {
        const int nl = 2 * wave + (r >> 3);
        const int d  = r & 7;
        const bf16_t* chp = &ch_lds[(nl * 8 + d) * 128];
#pragma unroll
        for (int s = 0; s < 4; ++s) {
            Frag fa;
            fa.u[0] = *(const uint4*)(chp + 32 * s + 8 * hi);
            fa.u[1] = *(const uint4*)(chp + 32 * s + 16 + 8 * hi);
            afrag[s] = fa.v;
        }
    }

    // per-lane a matrix for node g0n+hi, pre-paired over children:
    // ar2[cp*8+d] = { a[2cp][d], a[2cp+1][d] }  -> v_pk_fma_f32 contraction
    v2f ar2[32];
    {
        const float* ap = &a_lds[(2 * wave + hi) * 64];
#pragma unroll
        for (int cp = 0; cp < 4; ++cp)
#pragma unroll
            for (int d = 0; d < 8; ++d) {
                v2f t2v = { ap[(2 * cp) * 8 + d], ap[(2 * cp + 1) * 8 + d] };
                ar2[cp * 8 + d] = t2v;
            }
    }
    v2f mf2[4];
    {
        const unsigned char* mp = cmask_l + (size_t)(g0n + hi) * 8;
#pragma unroll
        for (int cp = 0; cp < 4; ++cp) {
            v2f m = { mp[2 * cp] ? 1.0f : 0.0f, mp[2 * cp + 1] ? 1.0f : 0.0f };
            mf2[cp] = m;
        }
    }

    // D-tile: lane holds column h=16t+r; VGPR v holds child row d=v of node g0n+hi.
    // g[c][h] = relu(sum_d a[c][d] * u[d][h]); pooled[h] = max over masked c (init 0).
    const v2f z2 = { 0.f, 0.f };
#pragma unroll
    for (int t = 0; t < 8; ++t) {
        v8f acc = {0.f, 0.f, 0.f, 0.f, 0.f, 0.f, 0.f, 0.f};
#pragma unroll
        for (int s = 0; s < 4; ++s) {
            Frag bb;
            const uint4* bp = (const uint4*)&wg_lds[((t * 4 + s) * 32 + lane) * 16];
            bb.u[0] = bp[0]; bb.u[1] = bp[1];
            acc = wmma_bf16(afrag[s], bb.v, acc);
        }
        v2f pv2 = z2;
#pragma unroll
        for (int cp = 0; cp < 4; ++cp) {
            v2f sum = z2;
#pragma unroll
            for (int d = 0; d < 8; ++d) {
                v2f u2 = { acc[d], acc[d] };
                sum += ar2[cp * 8 + d] * u2;            // v_pk_fma_f32
            }
            sum = __builtin_elementwise_max(sum, z2) * mf2[cp];   // relu * mask
            pv2 = __builtin_elementwise_max(pv2, sum);
        }
        float pv = fmaxf(pv2.x, pv2.y);
        pooled_lds[(2 * wave + hi) * 128 + t * 16 + r] = (bf16_t)pv;
    }
    __syncthreads();

    // ---- Phase 2: h = tanh(x @ W_ih + pooled @ W_hh + b). Wave owns N-tile t2=wave.
    const int t2 = wave;
    v16bf ax[4], apl[4];
    const int node_r = blk * 16 + r;
#pragma unroll
    for (int s = 0; s < 4; ++s) {
        Frag fx, fp;
#pragma unroll
        for (int c8 = 0; c8 < 2; ++c8) {
            int base = 32 * s + 16 * c8 + 8 * hi;
            const float4* xp = (const float4*)(x_l + (size_t)node_r * 128 + base);
            float4 f0 = xp[0], f1 = xp[1];
            Pack8 pk;
            pk.h[0] = (bf16_t)f0.x; pk.h[1] = (bf16_t)f0.y;
            pk.h[2] = (bf16_t)f0.z; pk.h[3] = (bf16_t)f0.w;
            pk.h[4] = (bf16_t)f1.x; pk.h[5] = (bf16_t)f1.y;
            pk.h[6] = (bf16_t)f1.z; pk.h[7] = (bf16_t)f1.w;
            fx.u[c8] = pk.u;
            fp.u[c8] = *(const uint4*)(&pooled_lds[r * 128 + base]);
        }
        ax[s]  = fx.v;
        apl[s] = fp.v;
    }

    v8f acc = {0.f, 0.f, 0.f, 0.f, 0.f, 0.f, 0.f, 0.f};
    const bf16_t* wih = wsw + 128 * 128;
    const bf16_t* whh = wsw + 2 * 128 * 128;
#pragma unroll
    for (int s = 0; s < 4; ++s) {
        Frag bb;
        const uint4* bp = (const uint4*)(wih + ((t2 * 4 + s) * 32 + lane) * 16);
        bb.u[0] = bp[0]; bb.u[1] = bp[1];
        acc = wmma_bf16(ax[s], bb.v, acc);
    }
#pragma unroll
    for (int s = 0; s < 4; ++s) {
        Frag bb;
        const uint4* bp = (const uint4*)(whh + ((t2 * 4 + s) * 32 + lane) * 16);
        bb.u[0] = bp[0]; bb.u[1] = bp[1];
        acc = wmma_bf16(apl[s], bb.v, acc);
    }

    float bias = b_ih[t2 * 16 + r] + b_hh[t2 * 16 + r];
#pragma unroll
    for (int v = 0; v < 8; ++v) {
        float o = fast_tanh(acc[v] + bias);
        h_out[(size_t)(blk * 16 + v + 8 * hi) * 128 + t2 * 16 + r] = o;
    }
}

// ---------------------------------------------------------------------------
extern "C" void kernel_launch(void* const* d_in, const int* in_sizes, int n_in,
                              void* d_out, int out_size, void* d_ws, size_t ws_size,
                              hipStream_t stream) {
    (void)in_sizes; (void)n_in; (void)out_size; (void)ws_size;

    const float*         node_features = (const float*)d_in[0];
    const int*           child_idx     = (const int*)d_in[1];
    const unsigned char* child_mask    = (const unsigned char*)d_in[2]; // jax bool = 1B
    const float*         A_c           = (const float*)d_in[3];
    const float*         W_g           = (const float*)d_in[4];
    const float*         W_ih          = (const float*)d_in[5];
    const float*         W_hh          = (const float*)d_in[6];
    const float*         b_ih          = (const float*)d_in[7];
    const float*         b_hh          = (const float*)d_in[8];
    float*               out           = (float*)d_out;

    // ws layout: [0,96KB) swizzled bf16 weights (3 x 32KB); then 8MB h ping buffer
    bf16_t* wsw = (bf16_t*)d_ws;
    float*  hY  = (float*)((char*)d_ws + 3 * 128 * 128 * sizeof(bf16_t));

    ran_prep_weights<<<12, 256, 0, stream>>>(W_g, W_ih, W_hh, wsw);

    // scan order: original level 5 first, level 0 last; last write must hit d_out
    for (int i = 0; i < L_LVL; ++i) {
        int li = L_LVL - 1 - i;
        const float* hin  = (i == 0) ? hY : (((i - 1) & 1) ? out : hY);
        float*       hout = (i & 1) ? out : hY;
        size_t off  = (size_t)li * N_NODES;
        ran_level_kernel<<<N_NODES / 16, 256, 0, stream>>>(
            node_features + off * H_DIM,
            child_idx + off * C_CH,
            child_mask + off * C_CH,
            A_c + off * C_CH * C_CH,
            wsw, b_ih, b_hh, hin, hout, (i == 0) ? 1 : 0);
    }
}
